// SelfAttentionWithGate_51694226374801
// MI455X (gfx1250) — matrix operand order repaired and checked by reading, not compiled, not run on hardware
//
#include <hip/hip_runtime.h>
#include <math.h>

// Problem constants (from the reference)
#define S_DIM 256
#define CQ    128
#define HC    128
#define NH    4
#define DH    32
#define NWAVE 16
#define NTHR  512

typedef __attribute__((ext_vector_type(16))) __bf16 v16bf;
typedef __attribute__((ext_vector_type(8)))  __bf16 v8bf;
typedef __attribute__((ext_vector_type(8)))  float  v8f;

// ---------------------------------------------------------------------------
// WMMA wrapper: D(f32 16x16) = A(bf16 16x32) * B(bf16 32x16) + C
// ---------------------------------------------------------------------------
__device__ __forceinline__ v8f wmma_bf16(v16bf a, v16bf b, v8f c) {
  return __builtin_amdgcn_wmma_f32_16x16x32_bf16(false, a, false, b,
                                                 (short)0, c, false, false);
}

// ---------------------------------------------------------------------------
// A-fragment (16x32 bf16, ISA 7.12.2): lane l holds row M=l%16,
// elements 0..7 -> K = 8*(l/16)+[0..8), elements 8..15 -> K = 8*(l/16)+16+[0..8)
// ---------------------------------------------------------------------------
__device__ __forceinline__ v16bf load_a_lds(const __bf16* base, int ld,
                                            int row, int k0, int hf) {
  union { v16bf v; v8bf h[2]; } u;
  const __bf16* p = base + row * ld + k0 + 8 * hf;
  u.h[0] = *(const v8bf*)(p);
  u.h[1] = *(const v8bf*)(p + 16);
  return u.v;
}

// Same A-fragment but sourced from global fp32 (convert to bf16 in-register).
__device__ __forceinline__ v16bf load_a_gf32(const float* base, int ld,
                                             int row, int k0, int hf) {
  const float* p = base + row * ld + k0 + 8 * hf;
  float4 f0 = *(const float4*)(p);
  float4 f1 = *(const float4*)(p + 4);
  float4 f2 = *(const float4*)(p + 16);
  float4 f3 = *(const float4*)(p + 20);
  v16bf v;
  v[0]  = (__bf16)f0.x; v[1]  = (__bf16)f0.y; v[2]  = (__bf16)f0.z; v[3]  = (__bf16)f0.w;
  v[4]  = (__bf16)f1.x; v[5]  = (__bf16)f1.y; v[6]  = (__bf16)f1.z; v[7]  = (__bf16)f1.w;
  v[8]  = (__bf16)f2.x; v[9]  = (__bf16)f2.y; v[10] = (__bf16)f2.z; v[11] = (__bf16)f2.w;
  v[12] = (__bf16)f3.x; v[13] = (__bf16)f3.y; v[14] = (__bf16)f3.z; v[15] = (__bf16)f3.w;
  return v;
}

// ---------------------------------------------------------------------------
// B-fragment (32x16 bf16): lane l holds column N=l%16, elements 0..7 -> K=8*(l/16)+[0..8),
// elements 8..15 -> K=8*(l/16)+16+[0..8). B stored transposed (M[n][k] row-major) -> two 16B reads per lane.
// ---------------------------------------------------------------------------
__device__ __forceinline__ v16bf load_bT_lds(const __bf16* base, int ld,
                                             int n, int k0, int hf) {
  union { v16bf v; v8bf h[2]; } u;
  const __bf16* p = base + n * ld + k0 + 8 * hf;
  u.h[0] = *(const v8bf*)(p);
  u.h[1] = *(const v8bf*)(p + 16);
  return u.v;
}

// ---------------------------------------------------------------------------
// Async copy of a 32 KB bf16 tile (global -> LDS) via the CDNA5 async-to-LDS
// path: 512 lanes x 16 B x 4 issues, tracked by ASYNCcnt.
// ---------------------------------------------------------------------------
__device__ __forceinline__ void async_copy_32k(const __bf16* __restrict__ gsrc,
                                               __bf16* lds_dst, int tid) {
  unsigned lbase = (unsigned)(size_t)lds_dst;   // low 32 bits = LDS offset
#pragma unroll
  for (int k = 0; k < 4; ++k) {
    unsigned off   = (unsigned)(tid * 16 + k * 8192);
    unsigned laddr = lbase + off;
    const char* g  = (const char*)gsrc + off;
    asm volatile("global_load_async_to_lds_b128 %0, %1, off"
                 :: "v"(laddr), "v"(g)
                 : "memory");
  }
  asm volatile("s_wait_asynccnt 0" ::: "memory");
}

// ---------------------------------------------------------------------------
// Prep kernel: convert W_qkvg (512x128) and W_o (128x128) fp32 -> bf16 in ws.
// ws layout: [0, 65536) = Wqkvg_bf, [65536, 81920) = Wo_bf   (bf16 elements)
// ---------------------------------------------------------------------------
__global__ __launch_bounds__(NTHR)
void cvt_weights_kernel(const float* __restrict__ Wq,
                        const float* __restrict__ Wo,
                        __bf16* __restrict__ dst) {
  int i = blockIdx.x * NTHR + threadIdx.x;   // 160 blocks * 512 = 81920
  float v = (i < 65536) ? Wq[i] : Wo[i - 65536];
  dst[i] = (__bf16)v;
}

// ---------------------------------------------------------------------------
// Fused gated self-attention: one workgroup (16 waves) per batch element.
// ---------------------------------------------------------------------------
__global__ __launch_bounds__(NTHR)
void attn_gate_kernel(const float* __restrict__ X,      // [256,256,128]
                      const float* __restrict__ mask,   // [256,256] (b, sk)
                      const float* __restrict__ bias,   // [4,256,256] (h,sq,sk)
                      const float* __restrict__ addt,   // [256,256,128]
                      const __bf16* __restrict__ Wbf,   // prepped bf16 weights
                      const float* __restrict__ gbias,  // [128]
                      const float* __restrict__ bo,     // [128]
                      float* __restrict__ out) {        // [256,256,128] = [s,b,c]
  extern __shared__ char smem[];
  __bf16* qs   = (__bf16*)(smem);            // [256][128] q*scale -> later gated o
  __bf16* kbuf = (__bf16*)(smem + 65536);    // [256][128]
  __bf16* vbT  = (__bf16*)(smem + 131072);   // [128][256]  (v transposed)
  __bf16* gb   = (__bf16*)(smem + 196608);   // [256][128]  sigmoid gate
  char*   scr  = smem + 262144;              // 32KB: W-tile / scores / partials / Wo
  __bf16* pb   = (__bf16*)(smem + 294912);   // [32][256] probs (bf16)

  const int b    = blockIdx.x;
  const int tid  = threadIdx.x;
  const int wid  = tid >> 5;        // 0..15
  const int lane = tid & 31;
  const int ln   = lane & 15;
  const int hf   = lane >> 4;

  const float SCALE = 0.17677669529663687f;  // 1/sqrt(32)
  const float* Xb = X + (size_t)b * S_DIM * CQ;

  // ======================= Phase 1: QKVG projection =======================
  v16bf ax[4];   // A-fragments of X, reused for all 4 W blocks
#pragma unroll
  for (int kc = 0; kc < 4; ++kc)
    ax[kc] = load_a_gf32(Xb, CQ, 16 * wid + ln, 32 * kc, hf);

  __bf16* Wt = (__bf16*)scr;  // [128][128] bf16 weight tile
  for (int f = 0; f < 4; ++f) {     // f: 0=q 1=k 2=v 3=g
    __syncthreads();                // scratch free from previous use
    async_copy_32k(Wbf + (size_t)f * 16384, Wt, tid);
    __syncthreads();
#pragma unroll
    for (int nt = 0; nt < 8; ++nt) {
      v8f acc = {};
#pragma unroll
      for (int kc = 0; kc < 4; ++kc)
        acc = wmma_bf16(ax[kc], load_bT_lds(Wt, 128, 16 * nt + ln, 32 * kc, hf), acc);
      const int j = 16 * nt + ln;
      if (f == 0) {
#pragma unroll
        for (int r = 0; r < 8; ++r) {
          int s = 16 * wid + r + 8 * hf;
          qs[s * HC + j] = (__bf16)(acc[r] * SCALE);
        }
      } else if (f == 1) {
#pragma unroll
        for (int r = 0; r < 8; ++r) {
          int s = 16 * wid + r + 8 * hf;
          kbuf[s * HC + j] = (__bf16)acc[r];
        }
      } else if (f == 2) {
        v8bf pk;
#pragma unroll
        for (int r = 0; r < 8; ++r) pk[r] = (__bf16)acc[r];
        *(v8bf*)(vbT + j * S_DIM + 16 * wid + 8 * hf) = pk;  // vbT[j][s..s+8)
      } else {
        float gj = gbias[j];
#pragma unroll
        for (int r = 0; r < 8; ++r) {
          int s = 16 * wid + r + 8 * hf;
          // fast sigmoid: v_rcp_f32 instead of IEEE divide
          gb[s * HC + j] = (__bf16)__builtin_amdgcn_rcpf(1.0f + __expf(-(acc[r] + gj)));
        }
      }
    }
  }
  __syncthreads();

  // ========================= Phase 2: attention ==========================
  float* scoreS = (float*)scr;   // [32][256] f32 scores / [4][32][32] partials
  for (int h = 0; h < NH; ++h) {
    for (int qb = 0; qb < 8; ++qb) {   // 32-query blocks
      // ---- scores: 2x16 tiles, 2 per wave, K = 32 (one WMMA) ----
      {
        const int mt  = wid >> 3;
        const int ntb = wid & 7;
#pragma unroll
        for (int t = 0; t < 2; ++t) {
          const int nt = ntb + 8 * t;
          v16bf aq = load_a_lds(qs,   HC, qb * 32 + 16 * mt + ln, 32 * h, hf);
          v16bf bk = load_bT_lds(kbuf, HC, 16 * nt + ln,          32 * h, hf);
          v8f acc = {};
          acc = wmma_bf16(aq, bk, acc);
          const int sk = 16 * nt + ln;
          const float mterm = (mask[b * S_DIM + sk] - 1.0f) * 1.0e9f;
#pragma unroll
          for (int r = 0; r < 8; ++r) {
            const int sql = 16 * mt + r + 8 * hf;     // 0..31
            const int sqg = qb * 32 + sql;
            scoreS[sql * S_DIM + sk] =
                acc[r] + bias[((size_t)h * S_DIM + sqg) * S_DIM + sk] + mterm;
          }
        }
      }
      __syncthreads();
      // ---- softmax: 2 rows per wave, full row in registers ----
#pragma unroll
      for (int rr = 0; rr < 2; ++rr) {
        const int row = 2 * wid + rr;
        float x[8];
#pragma unroll
        for (int k = 0; k < 8; ++k) x[k] = scoreS[row * S_DIM + lane + 32 * k];
        float m = x[0];
#pragma unroll
        for (int k = 1; k < 8; ++k) m = fmaxf(m, x[k]);
        for (int d = 1; d < 32; d <<= 1) m = fmaxf(m, __shfl_xor(m, d, 32));
        float sum = 0.0f;
#pragma unroll
        for (int k = 0; k < 8; ++k) { x[k] = __expf(x[k] - m); sum += x[k]; }
        for (int d = 1; d < 32; d <<= 1) sum += __shfl_xor(sum, d, 32);
        const float inv = __builtin_amdgcn_rcpf(sum);   // fast normalizer
#pragma unroll
        for (int k = 0; k < 8; ++k)
          pb[row * S_DIM + lane + 32 * k] = (__bf16)(x[k] * inv);
      }
      __syncthreads();
      // ---- P(32x256) * V(256x32): 4 tiles x split-K 4 over 16 waves ----
      {
        const int tile = wid & 3, seg = wid >> 2;
        const int mt2 = tile >> 1, nt2 = tile & 1;
        v8f acc = {};
#pragma unroll
        for (int kk = 0; kk < 2; ++kk) {
          const int k0 = seg * 64 + kk * 32;
          v16bf ap = load_a_lds(pb, S_DIM, 16 * mt2 + ln, k0, hf);
          v16bf bv = load_bT_lds(vbT, S_DIM, 32 * h + 16 * nt2 + ln, k0, hf);
          acc = wmma_bf16(ap, bv, acc);
        }
        float* part = scoreS;   // reuse scratch: [4][32][32]
#pragma unroll
        for (int r = 0; r < 8; ++r)
          part[seg * 1024 + (16 * mt2 + r + 8 * hf) * 32 + 16 * nt2 + ln] = acc[r];
      }
      __syncthreads();
      // ---- reduce split-K, apply sigmoid gate, store into qs (reuse) ----
      {
        float* part = scoreS;
#pragma unroll
        for (int e = 0; e < 2; ++e) {
          const int idx = e * NTHR + tid;     // 0..1023
          const int ql = idx >> 5, d = idx & 31;
          float s = part[ql * 32 + d] + part[1024 + ql * 32 + d] +
                    part[2048 + ql * 32 + d] + part[3072 + ql * 32 + d];
          const int sg = qb * 32 + ql;
          const float gate = (float)gb[sg * HC + 32 * h + d];
          qs[sg * HC + 32 * h + d] = (__bf16)(s * gate);
        }
      }
      __syncthreads();
    }
  }

  // ============ Phase 3: output projection + transpose + add =============
  __bf16* WoS = (__bf16*)scr;
  async_copy_32k(Wbf + 65536, WoS, tid);   // Wo_bf
  __syncthreads();
#pragma unroll
  for (int nt = 0; nt < 8; ++nt) {
    v8f acc = {};
#pragma unroll
    for (int kc = 0; kc < 4; ++kc) {
      v16bf ao = load_a_lds(qs,  HC, 16 * wid + ln, 32 * kc, hf);
      v16bf bw = load_bT_lds(WoS, HC, 16 * nt + ln, 32 * kc, hf);
      acc = wmma_bf16(ao, bw, acc);
    }
    const int oc = 16 * nt + ln;
    const float bov = bo[oc];
#pragma unroll
    for (int r = 0; r < 8; ++r) {
      const int s = 16 * wid + r + 8 * hf;
      // out[s][b][oc]; addt flat index (s*256+b)*128+oc matches after swapaxes
      const size_t gi = ((size_t)s * S_DIM + b) * CQ + oc;
      out[gi] = acc[r] + bov + addt[gi];
    }
  }
}

// ---------------------------------------------------------------------------
extern "C" void kernel_launch(void* const* d_in, const int* in_sizes, int n_in,
                              void* d_out, int out_size, void* d_ws, size_t ws_size,
                              hipStream_t stream) {
  const float* X    = (const float*)d_in[0];
  const float* mask = (const float*)d_in[1];
  const float* bias = (const float*)d_in[2];
  const float* addt = (const float*)d_in[3];
  const float* Wq   = (const float*)d_in[4];
  const float* gbia = (const float*)d_in[5];
  const float* Wo   = (const float*)d_in[6];
  const float* bo   = (const float*)d_in[7];
  float* out   = (float*)d_out;
  __bf16* wbf  = (__bf16*)d_ws;   // 81920 bf16 = 160 KB

  // One-time (per launch) weight conversion: 160 blocks x 512 = 81920 elems.
  cvt_weights_kernel<<<dim3(160), dim3(NTHR), 0, stream>>>(Wq, Wo, wbf);

  constexpr size_t SMEM = 304 * 1024;  // 4*64KB qkvg + 32KB scratch + 16KB probs
  attn_gate_kernel<<<dim3(256), dim3(NTHR), SMEM, stream>>>(
      X, mask, bias, addt, wbf, gbia, bo, out);
}
